// ProMolAtt_33500744909248
// MI455X (gfx1250) — matrix operand order repaired and checked by reading, not compiled, not run on hardware
//
#include <hip/hip_runtime.h>
#include <hip/hip_bf16.h>

typedef float  v8f   __attribute__((ext_vector_type(8)));
typedef __bf16 v8bf  __attribute__((ext_vector_type(8)));
typedef __bf16 v16bf __attribute__((ext_vector_type(16)));

union BF16x16 { v16bf v; v8bf h[2]; };

#define H        128
#define TILE_M   16
#define STRB     136   // bf16 row stride (elements); 272 B/row -> half-wave-conflict-free b128 reads

// ---- monotone uint encoding of f32 for atomicMax-based segment max ----
__device__ __forceinline__ unsigned enc_f32(float x) {
  unsigned u = __float_as_uint(x);
  return (u & 0x80000000u) ? ~u : (u | 0x80000000u);
}
__device__ __forceinline__ float dec_f32(unsigned e) {
  unsigned u = (e & 0x80000000u) ? (e ^ 0x80000000u) : ~e;
  return __uint_as_float(u);
}
#define ENC_NEG_INF 0x007FFFFFu   // enc(-inf)

// split f32 -> bf16 hi + bf16 lo (RNE), a ~= h + l with |err| <= 2^-16 |a|
__device__ __forceinline__ void split_bf16(float x, __bf16& h, __bf16& l) {
  h = (__bf16)x;
  l = (__bf16)(x - (float)h);
}

// Stage a 16 x 128 f32 tile into LDS as split bf16 hi/lo (256 threads).
__device__ __forceinline__ void stage_tile(const float* __restrict__ src,
                                           __bf16* sAh, __bf16* sAl, int tid) {
  int r  = tid >> 4;
  int c0 = (tid & 15) * 8;
  const float4* g = (const float4*)(src + (size_t)r * H + c0);
  float4 x0 = g[0];
  float4 x1 = g[1];
  float xs[8] = {x0.x, x0.y, x0.z, x0.w, x1.x, x1.y, x1.z, x1.w};
  v8bf h, l;
#pragma unroll
  for (int e = 0; e < 8; ++e) {
    __bf16 hh, ll;
    split_bf16(xs[e], hh, ll);
    h[e] = hh;
    l[e] = ll;
  }
  *(v8bf*)(sAh + r * STRB + c0) = h;   // 16B aligned: 272%16==0, c0*2%16==0
  *(v8bf*)(sAl + r * STRB + c0) = l;
}

// One wave computes a 16x16 tile of relu(A @ W^T + bias) via bf16x3 emulation:
// 4 K-chunks of 32, 3 x V_WMMA_F32_16X16X32_BF16 per chunk (12 WMMAs, K=128).
// vals[r] = row M = r + 8*(lane>=16), col cb + (lane&15).
__device__ __forceinline__ void wmma_tile(const __bf16* sAh, const __bf16* sAl,
                                          const __bf16* __restrict__ Wh,
                                          const __bf16* __restrict__ Wl,
                                          const float* __restrict__ bias,
                                          int cb, int lane, float vals[8]) {
  int n    = lane & 15;
  int half = lane >> 4;
  // ISA 16-bit A layout: lane m holds K = kc + half*8 + [0..7] and kc + 16 + half*8 + [0..7]
  const __bf16* arh = sAh + (lane & 15) * STRB + half * 8;
  const __bf16* arl = sAl + (lane & 15) * STRB + half * 8;
  // ISA 16-bit B layout: lane n holds K = kc + half*8 + [0..7] and kc + 16 + half*8 + [0..7] of column (cb+n) = row of W
  const __bf16* wrh = Wh + (size_t)(cb + n) * H + half * 8;
  const __bf16* wrl = Wl + (size_t)(cb + n) * H + half * 8;
  v8f c = {};
#pragma unroll
  for (int kc = 0; kc < H; kc += 32) {
    BF16x16 ah, al, bh, bl;
    ah.h[0] = *(const v8bf*)(arh + kc);
    ah.h[1] = *(const v8bf*)(arh + kc + 16);
    al.h[0] = *(const v8bf*)(arl + kc);
    al.h[1] = *(const v8bf*)(arl + kc + 16);
    bh.h[0] = *(const v8bf*)(wrh + kc);
    bh.h[1] = *(const v8bf*)(wrh + kc + 16);
    bl.h[0] = *(const v8bf*)(wrl + kc);
    bl.h[1] = *(const v8bf*)(wrl + kc + 16);
    c = __builtin_amdgcn_wmma_f32_16x16x32_bf16(false, ah.v, false, bh.v, (short)0, c, false, false);
    c = __builtin_amdgcn_wmma_f32_16x16x32_bf16(false, ah.v, false, bl.v, (short)0, c, false, false);
    c = __builtin_amdgcn_wmma_f32_16x16x32_bf16(false, al.v, false, bh.v, (short)0, c, false, false);
  }
  float bv = bias[cb + n];
#pragma unroll
  for (int r = 0; r < 8; ++r) vals[r] = fmaxf(c[r] + bv, 0.0f);
}

// reduce across the 16 lanes of each half-wave (masks < 16 stay in-half)
__device__ __forceinline__ float red16(float p) {
  p += __shfl_xor(p, 1, 32);
  p += __shfl_xor(p, 2, 32);
  p += __shfl_xor(p, 4, 32);
  p += __shfl_xor(p, 8, 32);
  return p;
}

// ---------------- kernel P: split Wq / Wk into bf16 hi/lo tables ----------------
__global__ void __launch_bounds__(256) k_prep(const float* __restrict__ Wq,
                                              const float* __restrict__ Wk,
                                              __bf16* qh, __bf16* ql,
                                              __bf16* kh, __bf16* kl) {
  int i = blockIdx.x * 256 + threadIdx.x;
  if (i < H * H) {
    __bf16 h, l;
    split_bf16(Wq[i], h, l);  qh[i] = h;  ql[i] = l;
    split_bf16(Wk[i], h, l);  kh[i] = h;  kl[i] = l;
  }
}

// ---------------- kernel 0: init accumulators ----------------
__global__ void __launch_bounds__(256) k_init(float* o, float* denom, unsigned* m_enc,
                                              int oN, int Bn) {
  int i = blockIdx.x * 256 + threadIdx.x;
  if (i < oN) o[i] = 0.0f;
  if (i < Bn) { denom[i] = 0.0f; m_enc[i] = ENC_NEG_INF; }
}

// ---------------- kernel 1: q = relu(mol@Wq^T+bq); qm[b] = q[b] . wm2 ----------------
__global__ void __launch_bounds__(256) k_q(const float* __restrict__ mol,
                                           const __bf16* __restrict__ Wqh,
                                           const __bf16* __restrict__ Wql,
                                           const float* __restrict__ bq,
                                           const float* __restrict__ Wm,
                                           float* __restrict__ q_out,
                                           float* __restrict__ qm) {
  __shared__ __bf16 sAh[TILE_M * STRB];
  __shared__ __bf16 sAl[TILE_M * STRB];
  __shared__ float  sQ[TILE_M];
  int tid  = threadIdx.x;
  int row0 = blockIdx.x * TILE_M;
  stage_tile(mol + (size_t)row0 * H, sAh, sAl, tid);
  if (tid < TILE_M) sQ[tid] = 0.0f;
  __syncthreads();

  int wave = tid >> 5, lane = tid & 31;
  int cb = wave * 16;
  float vals[8];
  wmma_tile(sAh, sAl, Wqh, Wql, bq, cb, lane, vals);

  int n = lane & 15, half = lane >> 4;
  float w = Wm[H + cb + n];                     // wm2
#pragma unroll
  for (int r = 0; r < 8; ++r) {
    int M = r + half * 8;
    q_out[(size_t)(row0 + M) * H + cb + n] = vals[r];
    float p = red16(vals[r] * w);
    if (n == 0) atomicAdd(&sQ[M], p);
  }
  __syncthreads();
  if (tid < TILE_M) qm[row0 + tid] = sQ[tid];
}

// ---------------- kernel 2: score[i] = relu(prot@Wk^T+bk)[i] . wm1 + qm[batch[i]] ----------------
__global__ void __launch_bounds__(256) k_score(const float* __restrict__ prot,
                                               const __bf16* __restrict__ Wkh,
                                               const __bf16* __restrict__ Wkl,
                                               const float* __restrict__ bk,
                                               const float* __restrict__ Wm,
                                               const int* __restrict__ batch,
                                               const float* __restrict__ qm,
                                               float* __restrict__ score) {
  __shared__ __bf16 sAh[TILE_M * STRB];
  __shared__ __bf16 sAl[TILE_M * STRB];
  __shared__ float  sS[TILE_M];
  int tid  = threadIdx.x;
  int row0 = blockIdx.x * TILE_M;
  stage_tile(prot + (size_t)row0 * H, sAh, sAl, tid);
  if (tid < TILE_M) sS[tid] = 0.0f;
  __syncthreads();

  int wave = tid >> 5, lane = tid & 31;
  int cb = wave * 16;
  float vals[8];
  wmma_tile(sAh, sAl, Wkh, Wkl, bk, cb, lane, vals);

  int n = lane & 15, half = lane >> 4;
  float w = Wm[cb + n];                         // wm1
#pragma unroll
  for (int r = 0; r < 8; ++r) {
    int M = r + half * 8;
    float p = red16(vals[r] * w);
    if (n == 0) atomicAdd(&sS[M], p);
  }
  __syncthreads();
  if (tid < TILE_M) {
    int i = row0 + tid;
    score[i] = sS[tid] + qm[batch[i]];
  }
}

// ---------------- kernel 3: segment max (LDS pre-reduced) ----------------
__global__ void __launch_bounds__(256) k_max(const float* __restrict__ score,
                                             const int* __restrict__ batch,
                                             unsigned* __restrict__ m_enc,
                                             int N, int Bn) {
  __shared__ unsigned sm[256];
  int t = threadIdx.x;
  sm[t] = 0u;
  __syncthreads();
  int i = blockIdx.x * 256 + t;
  if (i < N) atomicMax(&sm[batch[i]], enc_f32(score[i]));
  __syncthreads();
  if (t < Bn && sm[t] != 0u) atomicMax(&m_enc[t], sm[t]);
}

// ---------------- kernel 4: e = exp(score - m); denom = segment_sum(e) ----------------
__global__ void __launch_bounds__(256) k_exp(float* __restrict__ score,      // in: score, out: e
                                             const int* __restrict__ batch,
                                             const unsigned* __restrict__ m_enc,
                                             float* __restrict__ denom,
                                             int N, int Bn) {
  __shared__ float sd[256];
  int t = threadIdx.x;
  sd[t] = 0.0f;
  __syncthreads();
  int i = blockIdx.x * 256 + t;
  if (i < N) {
    int g = batch[i];
    float ex = __expf(score[i] - dec_f32(m_enc[g]));
    score[i] = ex;
    atomicAdd(&sd[g], ex);
  }
  __syncthreads();
  if (t < Bn && sd[t] != 0.0f) atomicAdd(&denom[t], sd[t]);
}

// ---------------- kernel 5: recompute k, o[b] += k[i] * e[i]/denom[b] ----------------
__global__ void __launch_bounds__(256) k_out(const float* __restrict__ prot,
                                             const __bf16* __restrict__ Wkh,
                                             const __bf16* __restrict__ Wkl,
                                             const float* __restrict__ bk,
                                             const int* __restrict__ batch,
                                             const float* __restrict__ e,
                                             const float* __restrict__ denom,
                                             float* __restrict__ o) {
  __shared__ __bf16 sAh[TILE_M * STRB];
  __shared__ __bf16 sAl[TILE_M * STRB];
  __shared__ float  sAttn[TILE_M];
  __shared__ int    sB[TILE_M];
  int tid  = threadIdx.x;
  int row0 = blockIdx.x * TILE_M;
  stage_tile(prot + (size_t)row0 * H, sAh, sAl, tid);
  if (tid < TILE_M) {
    int i = row0 + tid;
    int g = batch[i];
    sB[tid]    = g;
    sAttn[tid] = e[i] / denom[g];
  }
  __syncthreads();

  int wave = tid >> 5, lane = tid & 31;
  int cb = wave * 16;
  float vals[8];
  wmma_tile(sAh, sAl, Wkh, Wkl, bk, cb, lane, vals);

  int n = lane & 15, half = lane >> 4;
  int col  = cb + n;
  int curg = sB[half * 8];
  float acc = 0.0f;
#pragma unroll
  for (int r = 0; r < 8; ++r) {          // sorted batch: rows mostly share a graph
    int M = r + half * 8;
    int g = sB[M];
    float contrib = vals[r] * sAttn[M];
    if (g != curg) {
      atomicAdd(&o[(size_t)curg * H + col], acc);
      curg = g; acc = 0.0f;
    }
    acc += contrib;
  }
  atomicAdd(&o[(size_t)curg * H + col], acc);
}

extern "C" void kernel_launch(void* const* d_in, const int* in_sizes, int n_in,
                              void* d_out, int out_size, void* d_ws, size_t ws_size,
                              hipStream_t stream) {
  const float* mol   = (const float*)d_in[0];
  const float* prot  = (const float*)d_in[1];
  const int*   batch = (const int*)d_in[2];
  const float* Wq    = (const float*)d_in[3];
  const float* bq    = (const float*)d_in[4];
  const float* Wk    = (const float*)d_in[5];
  const float* bk    = (const float*)d_in[6];
  const float* Wm    = (const float*)d_in[7];

  int Bn = in_sizes[0] / H;          // 256
  int N  = in_sizes[1] / H;          // 262144

  float* out = (float*)d_out;
  float* o   = out;                  // [B,H]
  float* q   = out + (size_t)Bn * H; // [B,H]

  // workspace layout (~1.2 MB)
  float*    score = (float*)d_ws;                 // N floats (reused as e)
  float*    qm    = score + N;                    // Bn
  float*    denom = qm + Bn;                      // Bn
  unsigned* m_enc = (unsigned*)(denom + Bn);      // Bn
  __bf16*   Wqh   = (__bf16*)(m_enc + Bn);        // H*H bf16 each
  __bf16*   Wql   = Wqh + H * H;
  __bf16*   Wkh   = Wql + H * H;
  __bf16*   Wkl   = Wkh + H * H;

  int oN = Bn * H;
  k_prep <<<(H * H + 255) / 256, 256, 0, stream>>>(Wq, Wk, Wqh, Wql, Wkh, Wkl);
  k_init <<<(oN + 255) / 256,    256, 0, stream>>>(o, denom, m_enc, oN, Bn);
  k_q    <<<Bn / TILE_M,         256, 0, stream>>>(mol, Wqh, Wql, bq, Wm, q, qm);
  k_score<<<N / TILE_M,          256, 0, stream>>>(prot, Wkh, Wkl, bk, Wm, batch, qm, score);
  k_max  <<<(N + 255) / 256,     256, 0, stream>>>(score, batch, m_enc, N, Bn);
  k_exp  <<<(N + 255) / 256,     256, 0, stream>>>(score, batch, m_enc, denom, N, Bn);
  k_out  <<<N / TILE_M,          256, 0, stream>>>(prot, Wkh, Wkl, bk, batch, score, denom, o);
}
